// TimeMoeSparseExpertsLayer_13580686590208
// MI455X (gfx1250) — compile-verified, matrix operand order checked by reading, not run on hardware
//
#include <hip/hip_runtime.h>
#include <math.h>

typedef __attribute__((ext_vector_type(4)))  float  f4;
typedef __attribute__((ext_vector_type(8)))  float  v8f;
typedef __attribute__((ext_vector_type(16))) __bf16 v16bf;

#define DM 1024      // hidden size
#define E_EXP 8      // experts
#define APITCH 40    // halves per LDS A row (32 K + 8 pad) = 80 B, b128-aligned
#define BPITCH 40    // halves per LDS B row (indexed by n, K contiguous)

// GEMM tiling: 256 threads = 8 waves; workgroup tile 64(M) x 128(N); K-block 32.
// Each wave owns a 32x32 output tile = 2x2 WMMA 16x16 accumulators.
// fp32 emulated via split-bf16 (bf16x3): x = hi + lo, x*w ~= hi*hi + hi*lo + lo*hi.

union Frag16 { v16bf v; f4 f[2]; };
union Pack8  { __bf16 h[8]; f4 f; };
union Pack2  { __bf16 h[2]; unsigned u; };

__device__ __forceinline__ v8f wmma_bf16(v16bf a, v16bf b, v8f c) {
  // D(16x16 f32) = A(16x32 bf16) * B(32x16 bf16) + C
  return __builtin_amdgcn_wmma_f32_16x16x32_bf16(false, a, false, b, (short)0, c,
                                                 false, false);
}

__device__ __forceinline__ void split_f32(float x, __bf16& h, __bf16& l) {
  const __bf16 hh = (__bf16)x;
  h = hh;
  l = (__bf16)(x - (float)hh);
}

// Load one 16x32 bf16 fragment row-slice for this lane.
// ISA layout: VGPRs 0-3 hold K = 8*kp .. 8*kp+7, VGPRs 4-7 hold K = 16+8*kp .. 16+8*kp+7.
__device__ __forceinline__ v16bf ld_frag(const __bf16* rowbase, int kp) {
  Frag16 r;
  r.f[0] = *(const f4*)(rowbase + kp * 8);
  r.f[1] = *(const f4*)(rowbase + 16 + kp * 8);
  return r.v;
}

// Stage a 2x8 (k x n) chunk of a row-major f32 weight matrix into transposed
// bf16 hi/lo LDS tiles [n][k]; adjacent-k pairs packed into one b32 store.
__device__ __forceinline__ void stage_b(const float* __restrict__ W, int ldW,
                                        int k0, int col0, int kr, int nc,
                                        __bf16* Bh, __bf16* Bl) {
  const float* r0 = W + (size_t)(k0 + kr) * ldW + col0 + nc;
  const float* r1 = r0 + ldW;
  const f4 x0 = *(const f4*)r0;
  const f4 x1 = *(const f4*)(r0 + 4);
  const f4 y0 = *(const f4*)r1;
  const f4 y1 = *(const f4*)(r1 + 4);
  const float v0[8] = {x0.x, x0.y, x0.z, x0.w, x1.x, x1.y, x1.z, x1.w};
  const float v1[8] = {y0.x, y0.y, y0.z, y0.w, y1.x, y1.y, y1.z, y1.w};
#pragma unroll
  for (int c = 0; c < 8; ++c) {
    Pack2 hp, lp;
    split_f32(v0[c], hp.h[0], lp.h[0]);
    split_f32(v1[c], hp.h[1], lp.h[1]);
    *(unsigned*)&Bh[(size_t)(nc + c) * BPITCH + kr] = hp.u;  // kr even -> 4B aligned
    *(unsigned*)&Bl[(size_t)(nc + c) * BPITCH + kr] = lp.u;
  }
}

// ---------------------------------------------------------------------------
// Router: logits = x @ gate_w (T x 8), softmax -> top2 -> normalized combine.
// Also shared_gate = sigmoid(x @ shared_gate_w). One wave per token.
// ---------------------------------------------------------------------------
__global__ __launch_bounds__(256)
void moe_router_kernel(const float* __restrict__ X,
                       const float* __restrict__ GW,
                       const float* __restrict__ SGW,
                       float* __restrict__ combine,
                       float* __restrict__ sgate,
                       float* __restrict__ logits,
                       int T) {
  const int lane  = threadIdx.x & 31;
  const int token = blockIdx.x * 8 + (threadIdx.x >> 5);
  if (token >= T) return;
  const float* xr = X + (size_t)token * DM;

  float acc[9];
#pragma unroll
  for (int i = 0; i < 9; ++i) acc[i] = 0.f;
  for (int d = lane; d < DM; d += 32) {
    const float xv = xr[d];
    const float* g = GW + (size_t)d * E_EXP;
#pragma unroll
    for (int e = 0; e < E_EXP; ++e) acc[e] += xv * g[e];
    acc[8] += xv * SGW[d];
  }
#pragma unroll
  for (int off = 16; off > 0; off >>= 1) {
#pragma unroll
    for (int i = 0; i < 9; ++i) acc[i] += __shfl_xor(acc[i], off, 32);
  }
  if (lane == 0) {
    float mx = acc[0];
#pragma unroll
    for (int e = 1; e < E_EXP; ++e) mx = fmaxf(mx, acc[e]);
    float p[E_EXP];
    float s = 0.f;
#pragma unroll
    for (int e = 0; e < E_EXP; ++e) { p[e] = __expf(acc[e] - mx); s += p[e]; }
    const float inv = 1.f / s;
#pragma unroll
    for (int e = 0; e < E_EXP; ++e) p[e] *= inv;
    int i1 = 0;
#pragma unroll
    for (int e = 1; e < E_EXP; ++e) if (p[e] > p[i1]) i1 = e;
    int i2 = -1;
    float p2 = -1.f;
#pragma unroll
    for (int e = 0; e < E_EXP; ++e)
      if (e != i1 && p[e] > p2) { p2 = p[e]; i2 = e; }
    const float w = fmaxf(p[i1] + p2, 1e-6f);
    float* c = combine + (size_t)token * E_EXP;
#pragma unroll
    for (int e = 0; e < E_EXP; ++e) c[e] = 0.f;
    c[i1] = p[i1] / w;
    c[i2] = p2 / w;
    sgate[token] = 1.f / (1.f + __expf(-acc[8]));
    float* lo = logits + (size_t)token * E_EXP;
#pragma unroll
    for (int e = 0; e < E_EXP; ++e) lo[e] = acc[e];
  }
}

// ---------------------------------------------------------------------------
// Fused gate/up GEMM (bf16x3): H = silu(X @ Wg) * (X @ Wu), H emitted as
// split bf16 hi/lo planes so the down GEMM needs no reconversion.
// ---------------------------------------------------------------------------
__global__ __launch_bounds__(256)
void moe_gateup_kernel(const float* __restrict__ X,
                       const float* __restrict__ Wg,
                       const float* __restrict__ Wu,
                       __bf16* __restrict__ Hh,
                       __bf16* __restrict__ Hl,
                       int K, int ldW) {
  __shared__ __bf16 Ah[64 * APITCH], Al[64 * APITCH];
  __shared__ __bf16 Bgh[128 * BPITCH], Bgl[128 * BPITCH];
  __shared__ __bf16 Buh[128 * BPITCH], Bul[128 * BPITCH];

  const int tid  = threadIdx.x;
  const int lane = tid & 31;
  const int wid  = tid >> 5;
  const int row0 = blockIdx.x * 64;
  const int col0 = blockIdx.y * 128;
  const int wr   = (wid >> 2) * 32;
  const int wc   = (wid & 3) * 32;
  const int m16  = lane & 15;
  const int kp   = lane >> 4;

  v8f cg[2][2], cu[2][2];
  const v8f vz = {};
#pragma unroll
  for (int i = 0; i < 2; ++i)
#pragma unroll
    for (int j = 0; j < 2; ++j) { cg[i][j] = vz; cu[i][j] = vz; }

  const int am  = tid >> 2;          // A row 0..63
  const int akc = (tid & 3) * 8;     // 8-wide K chunk
  const int bkr = (tid >> 4) * 2;    // B row pair 0,2,..,30
  const int bnc = (tid & 15) * 8;    // 8-wide N chunk

  for (int k0 = 0; k0 < K; k0 += 32) {
    {  // stage A 64x32 -> bf16 hi/lo [m][k]
      const float* ap = X + (size_t)(row0 + am) * K + k0 + akc;
      const f4 a0 = *(const f4*)ap;
      const f4 a1 = *(const f4*)(ap + 4);
      const float v[8] = {a0.x, a0.y, a0.z, a0.w, a1.x, a1.y, a1.z, a1.w};
      Pack8 ph, pl;
#pragma unroll
      for (int c = 0; c < 8; ++c) split_f32(v[c], ph.h[c], pl.h[c]);
      *(f4*)&Ah[am * APITCH + akc] = ph.f;
      *(f4*)&Al[am * APITCH + akc] = pl.f;
    }
    stage_b(Wg, ldW, k0, col0, bkr, bnc, Bgh, Bgl);
    stage_b(Wu, ldW, k0, col0, bkr, bnc, Buh, Bul);
    if (k0 + 32 < K) {  // prefetch next K block (global_prefetch_b8)
      __builtin_prefetch(X + (size_t)(row0 + am) * K + k0 + 32 + akc, 0, 1);
      __builtin_prefetch(Wg + (size_t)(k0 + 32 + bkr) * ldW + col0 + bnc, 0, 1);
      __builtin_prefetch(Wu + (size_t)(k0 + 32 + bkr) * ldW + col0 + bnc, 0, 1);
    }
    __syncthreads();

    v16bf ah[2], al2[2];
#pragma unroll
    for (int i = 0; i < 2; ++i) {
      const int rb = (wr + 16 * i + m16) * APITCH;
      ah[i]  = ld_frag(&Ah[rb], kp);
      al2[i] = ld_frag(&Al[rb], kp);
    }
#pragma unroll
    for (int j = 0; j < 2; ++j) {
      const int cb = (wc + 16 * j + m16) * BPITCH;
      const v16bf gh = ld_frag(&Bgh[cb], kp);
      const v16bf gl = ld_frag(&Bgl[cb], kp);
      const v16bf uh = ld_frag(&Buh[cb], kp);
      const v16bf ul = ld_frag(&Bul[cb], kp);
#pragma unroll
      for (int i = 0; i < 2; ++i) {
        cg[i][j] = wmma_bf16(al2[i], gh, cg[i][j]);  // lo*hi
        cg[i][j] = wmma_bf16(ah[i],  gl, cg[i][j]);  // hi*lo
        cg[i][j] = wmma_bf16(ah[i],  gh, cg[i][j]);  // hi*hi
        cu[i][j] = wmma_bf16(al2[i], uh, cu[i][j]);
        cu[i][j] = wmma_bf16(ah[i],  ul, cu[i][j]);
        cu[i][j] = wmma_bf16(ah[i],  uh, cu[i][j]);
      }
    }
    __syncthreads();
  }

  // epilogue: H = silu(g) * u, split into bf16 hi/lo planes
#pragma unroll
  for (int i = 0; i < 2; ++i)
#pragma unroll
    for (int j = 0; j < 2; ++j)
#pragma unroll
      for (int r = 0; r < 8; ++r) {
        const int row = row0 + wr + 16 * i + r + 8 * kp;
        const int col = col0 + wc + 16 * j + m16;
        const float g = cg[i][j][r];
        const float u = cu[i][j][r];
        const float sig = 1.f / (1.f + __expf(-g));
        const float h = g * sig * u;
        __bf16 hh, hl;
        split_f32(h, hh, hl);
        Hh[(size_t)row * DM + col] = hh;
        Hl[(size_t)row * DM + col] = hl;
      }
}

// ---------------------------------------------------------------------------
// Down GEMM (bf16x3) with per-row scaling and optional accumulation:
//   Out[row, col] (+)= scale[row*sstride+soff] * (H @ W)[row, col]
// H given as bf16 hi/lo planes [T, 1024]; W row-major f32 [K, DM].
// ---------------------------------------------------------------------------
__global__ __launch_bounds__(256)
void moe_down_kernel(const __bf16* __restrict__ Hh,
                     const __bf16* __restrict__ Hl,
                     const float* __restrict__ W,
                     float* __restrict__ Out,
                     int K, int ldW,
                     const float* __restrict__ scale,
                     int sstride, int soff, int beta) {
  __shared__ __bf16 Ah[64 * APITCH], Al[64 * APITCH];
  __shared__ __bf16 Bh[128 * BPITCH], Bl[128 * BPITCH];

  const int tid  = threadIdx.x;
  const int lane = tid & 31;
  const int wid  = tid >> 5;
  const int row0 = blockIdx.x * 64;
  const int col0 = blockIdx.y * 128;
  const int wr   = (wid >> 2) * 32;
  const int wc   = (wid & 3) * 32;
  const int m16  = lane & 15;
  const int kp   = lane >> 4;

  v8f c[2][2];
  const v8f vz = {};
#pragma unroll
  for (int i = 0; i < 2; ++i)
#pragma unroll
    for (int j = 0; j < 2; ++j) c[i][j] = vz;

  const int am  = tid >> 2;
  const int akc = (tid & 3) * 8;
  const int bkr = (tid >> 4) * 2;
  const int bnc = (tid & 15) * 8;

  for (int k0 = 0; k0 < K; k0 += 32) {
    // stage A: straight 16B copies of the pre-split bf16 H planes
    *(f4*)&Ah[am * APITCH + akc] =
        *(const f4*)(Hh + (size_t)(row0 + am) * DM + k0 + akc);
    *(f4*)&Al[am * APITCH + akc] =
        *(const f4*)(Hl + (size_t)(row0 + am) * DM + k0 + akc);
    stage_b(W, ldW, k0, col0, bkr, bnc, Bh, Bl);
    if (k0 + 32 < K) {
      __builtin_prefetch(Hh + (size_t)(row0 + am) * DM + k0 + 32 + akc, 0, 1);
      __builtin_prefetch(W + (size_t)(k0 + 32 + bkr) * ldW + col0 + bnc, 0, 1);
    }
    __syncthreads();

    v16bf ah[2], al2[2];
#pragma unroll
    for (int i = 0; i < 2; ++i) {
      const int rb = (wr + 16 * i + m16) * APITCH;
      ah[i]  = ld_frag(&Ah[rb], kp);
      al2[i] = ld_frag(&Al[rb], kp);
    }
#pragma unroll
    for (int j = 0; j < 2; ++j) {
      const int cb = (wc + 16 * j + m16) * BPITCH;
      const v16bf bh = ld_frag(&Bh[cb], kp);
      const v16bf bl = ld_frag(&Bl[cb], kp);
#pragma unroll
      for (int i = 0; i < 2; ++i) {
        c[i][j] = wmma_bf16(al2[i], bh, c[i][j]);
        c[i][j] = wmma_bf16(ah[i],  bl, c[i][j]);
        c[i][j] = wmma_bf16(ah[i],  bh, c[i][j]);
      }
    }
    __syncthreads();
  }

#pragma unroll
  for (int i = 0; i < 2; ++i)
#pragma unroll
    for (int j = 0; j < 2; ++j)
#pragma unroll
      for (int r = 0; r < 8; ++r) {
        const int row = row0 + wr + 16 * i + r + 8 * kp;
        const int col = col0 + wc + 16 * j + m16;
        const float s = scale[(size_t)row * sstride + soff];
        float v = s * c[i][j][r];
        float* o = Out + (size_t)row * DM + col;
        if (beta) v += *o;
        *o = v;
      }
}

// ---------------------------------------------------------------------------
extern "C" void kernel_launch(void* const* d_in, const int* in_sizes, int n_in,
                              void* d_out, int out_size, void* d_ws,
                              size_t ws_size, hipStream_t stream) {
  (void)n_in; (void)out_size; (void)ws_size;
  const float* x   = (const float*)d_in[0];  // [T, 1024]
  const float* gw  = (const float*)d_in[1];  // [1024, 8]
  const float* eg  = (const float*)d_in[2];  // [8, 1024, 1024]
  const float* eu  = (const float*)d_in[3];  // [8, 1024, 1024]
  const float* ed  = (const float*)d_in[4];  // [8, 1024, 1024]
  const float* sg  = (const float*)d_in[5];  // [1024, 4096]
  const float* su  = (const float*)d_in[6];  // [1024, 4096]
  const float* sd  = (const float*)d_in[7];  // [4096, 1024]
  const float* sgw = (const float*)d_in[8];  // [1024, 1]

  float* out = (float*)d_out;                // [T*1024] then [T*8] logits
  const int T = in_sizes[0] / DM;            // 8192
  float* logits = out + (size_t)T * DM;

  // workspace: Hh [T,1024] bf16 | Hl [T,1024] bf16 | combine [T,8] | sgate [T]
  __bf16* Hh = (__bf16*)d_ws;
  __bf16* Hl = Hh + (size_t)T * DM;
  float* combine = (float*)(Hl + (size_t)T * DM);
  float* sgatev  = combine + (size_t)T * E_EXP;

  moe_router_kernel<<<dim3((T + 7) / 8), dim3(256), 0, stream>>>(
      x, gw, sgw, combine, sgatev, logits, T);

  const dim3 grid(T / 64, 1024 / 128);
  const dim3 blk(256);

  // routed experts (dense, weighted by combine; expert 0 initializes out)
  for (int e = 0; e < E_EXP; ++e) {
    const float* wg_e = eg + (size_t)e * DM * 1024;
    const float* wu_e = eu + (size_t)e * DM * 1024;
    const float* wd_e = ed + (size_t)e * 1024 * DM;
    moe_gateup_kernel<<<grid, blk, 0, stream>>>(x, wg_e, wu_e, Hh, Hl, DM, 1024);
    moe_down_kernel<<<grid, blk, 0, stream>>>(Hh, Hl, wd_e, out, 1024, DM,
                                              combine, E_EXP, e, (e > 0) ? 1 : 0);
  }

  // shared expert (I_SH = 4096) in 4 column chunks reusing the L2-resident H
  for (int ks = 0; ks < 4; ++ks) {
    moe_gateup_kernel<<<grid, blk, 0, stream>>>(
        x, sg + (size_t)ks * 1024, su + (size_t)ks * 1024, Hh, Hl, DM, 4096);
    moe_down_kernel<<<grid, blk, 0, stream>>>(
        Hh, Hl, sd + (size_t)ks * 1024 * DM, out, 1024, DM, sgatev, 1, 0, 1);
  }
}
